// Jurassic3Attention_59004260712805
// MI455X (gfx1250) — matrix operand order repaired and checked by reading, not compiled, not run on hardware
//
#include <hip/hip_runtime.h>

// ---------------------------------------------------------------------------
// Jurassic3 attention block for MI455X (gfx1250, wave32, WMMA).
//   1. f32 -> bf16 conversion; weights transposed to [N][K] so WMMA
//      B-fragments are contiguous.
//   2. QKV GEMM (4096x6144x4096) bf16 WMMA, async-LDS double-buffered.
//   3. NeoX RoPE in-place on Q/K heads (bf16).
//   4. Flash attention: 4 waves/block share async-staged K + transposed V
//      LDS tiles; 16x16x32 bf16 WMMA for Q.K^T and P.V; online softmax.
//   5. O GEMM (4096x4096x4096) bf16 WMMA -> f32 d_out.
// ---------------------------------------------------------------------------

#define S_LEN   2048
#define BATCH   2
#define HIDDEN  4096
#define NH      32
#define NKV     8
#define HD      128
#define QKV_LD  6144
#define M_ROWS  (BATCH * S_LEN)   // 4096

typedef __attribute__((ext_vector_type(16))) __bf16 v16bf;
typedef __attribute__((ext_vector_type(8)))  float  v8f;
typedef int v4i __attribute__((vector_size(16)));   // matches builtin param type

#define GLOBAL_AS __attribute__((address_space(1)))
#define LDS_AS    __attribute__((address_space(3)))

__device__ __forceinline__ unsigned short f2bf(float f) {
  unsigned u = __builtin_bit_cast(unsigned, f);
  u += 0x7FFFu + ((u >> 16) & 1u);
  return (unsigned short)(u >> 16);
}
__device__ __forceinline__ float bf2f(unsigned short h) {
  unsigned u = ((unsigned)h) << 16;
  return __builtin_bit_cast(float, u);
}

union BF16Frag {
  v16bf v;
  uint4 q[2];
  unsigned short s[16];
};

__device__ __forceinline__ v8f wmma_bf16(const BF16Frag& a, const BF16Frag& b, v8f c) {
  return __builtin_amdgcn_wmma_f32_16x16x32_bf16(false, a.v, false, b.v,
                                                 (short)0, c, false, false);
}

// ---- async global->LDS copy (ASYNCcnt path), with synchronous fallback ----

__device__ __forceinline__ void async_copy16(const unsigned short* g, unsigned short* l) {
#if __has_builtin(__builtin_amdgcn_global_load_async_to_lds_b128)
  __builtin_amdgcn_global_load_async_to_lds_b128(
      (GLOBAL_AS v4i*)(g),
      (LDS_AS v4i*)(l), 0, 0);
#else
  *(uint4*)l = *(const uint4*)g;
#endif
}

template <int N>
__device__ __forceinline__ void wait_async() {
#if __has_builtin(__builtin_amdgcn_s_wait_asynccnt)
  __builtin_amdgcn_s_wait_asynccnt(N);
#elif defined(__AMDGCN__)
  asm volatile("s_wait_asynccnt %0" ::"n"(N) : "memory");
#endif
}

// --------------------------- conversion kernels ----------------------------

__global__ __launch_bounds__(256)
void cvt_bf16_kernel(const float* __restrict__ in, unsigned short* __restrict__ out, int n4) {
  int i = blockIdx.x * blockDim.x + threadIdx.x;
  if (i >= n4) return;
  float4 f = ((const float4*)in)[i];
  union { unsigned short s[4]; uint2 u; } r;
  r.s[0] = f2bf(f.x); r.s[1] = f2bf(f.y); r.s[2] = f2bf(f.z); r.s[3] = f2bf(f.w);
  ((uint2*)out)[i] = r.u;
}

// in: [K][N] f32 row-major  ->  out: [N][K] bf16 row-major (tiled transpose)
__global__ __launch_bounds__(256)
void cvt_transpose_kernel(const float* __restrict__ in, unsigned short* __restrict__ out,
                          int K, int N) {
  __shared__ float tile[32][33];
  int k0 = blockIdx.y * 32, n0 = blockIdx.x * 32;
  int tx = threadIdx.x & 31, ty = threadIdx.x >> 5;
#pragma unroll
  for (int r = ty; r < 32; r += 8)
    tile[r][tx] = in[(size_t)(k0 + r) * N + n0 + tx];
  __syncthreads();
#pragma unroll
  for (int r = ty; r < 32; r += 8)
    out[(size_t)(n0 + r) * K + k0 + tx] = f2bf(tile[tx][r]);
}

// ------------------------------- bf16 GEMM ---------------------------------
// C[M][N] = A[M][K] * B[K][N] with B pre-transposed (BT[N][K]).
// Block = 256 threads = 8 waves; block tile 256(M) x 128(N); wave tile 64x64.
// K-step 32; async global->LDS double-buffered staging (6 async/thread/stage).

template <bool F32OUT>
__global__ __launch_bounds__(256)
void gemm_bf16(const unsigned short* __restrict__ A,
               const unsigned short* __restrict__ BT,
               void* __restrict__ Cout, int M, int N, int K) {
  __shared__ unsigned short Abuf[2][256 * 32];   // 32 KB
  __shared__ unsigned short Bbuf[2][128 * 32];   // 16 KB

  const int tid  = threadIdx.x;
  const int lane = tid & 31;
  const int wave = tid >> 5;
  const int half = lane >> 4;
  const int l16  = lane & 15;

  const int bm = blockIdx.y * 256;
  const int bn = blockIdx.x * 128;
  const int wmL = (wave >> 1) * 64;   // 4 M-groups of 64
  const int wnL = (wave & 1) * 64;    // 2 N-groups of 64

  const int r0 = tid >> 2, cc = tid & 3;   // staging role: 4 threads per row

  v8f acc[4][4];
#pragma unroll
  for (int mi = 0; mi < 4; ++mi)
#pragma unroll
    for (int ni = 0; ni < 4; ++ni)
#pragma unroll
      for (int r = 0; r < 8; ++r) acc[mi][ni][r] = 0.f;

  const int NS = K >> 5;   // stages of 32 K

  auto issue_stage = [&](int s, int bufi) {
    int k0 = s * 32;
    const unsigned short* gA = A + (size_t)(bm + r0) * K + k0 + cc * 8;
    unsigned short* lA = &Abuf[bufi][r0 * 32 + cc * 8];
#pragma unroll
    for (int j = 0; j < 4; ++j)          // A: 256 rows
      async_copy16(gA + (size_t)j * 64 * K, lA + j * 64 * 32);
    const unsigned short* gB = BT + (size_t)(bn + r0) * K + k0 + cc * 8;
    unsigned short* lB = &Bbuf[bufi][r0 * 32 + cc * 8];
#pragma unroll
    for (int j = 0; j < 2; ++j)          // B: 128 rows
      async_copy16(gB + (size_t)j * 64 * K, lB + j * 64 * 32);
  };

  issue_stage(0, 0);

  for (int s = 0; s < NS; ++s) {
    if (s + 1 < NS) {
      issue_stage(s + 1, (s + 1) & 1);
      wait_async<6>();     // stage s (issued earlier) complete for this wave
    } else {
      wait_async<0>();
    }
    __syncthreads();       // stage s visible to all waves

    const unsigned short* As = &Abuf[s & 1][0];
    const unsigned short* Bs = &Bbuf[s & 1][0];
    BF16Frag a[4], b[4];
#pragma unroll
    for (int mi = 0; mi < 4; ++mi) {
      const unsigned short* p = As + (wmL + mi * 16 + l16) * 32;
      a[mi].q[0] = *(const uint4*)(p + half * 8);
      a[mi].q[1] = *(const uint4*)(p + 16 + half * 8);
    }
#pragma unroll
    for (int ni = 0; ni < 4; ++ni) {
      const unsigned short* p = Bs + (wnL + ni * 16 + l16) * 32;
      b[ni].q[0] = *(const uint4*)(p + half * 8);
      b[ni].q[1] = *(const uint4*)(p + 16 + half * 8);
    }
#pragma unroll
    for (int mi = 0; mi < 4; ++mi)
#pragma unroll
      for (int ni = 0; ni < 4; ++ni)
        acc[mi][ni] = wmma_bf16(a[mi], b[ni], acc[mi][ni]);

    __syncthreads();       // everyone done reading buf[s&1] before overwrite
  }

#pragma unroll
  for (int mi = 0; mi < 4; ++mi)
#pragma unroll
    for (int ni = 0; ni < 4; ++ni)
#pragma unroll
      for (int r = 0; r < 8; ++r) {
        int row = bm + wmL + mi * 16 + r + 8 * half;
        int col = bn + wnL + ni * 16 + l16;
        float v = acc[mi][ni][r];
        if (F32OUT) ((float*)Cout)[(size_t)row * N + col] = v;
        else        ((unsigned short*)Cout)[(size_t)row * N + col] = f2bf(v);
      }
}

// -------------------------------- RoPE -------------------------------------

__global__ __launch_bounds__(256)
void rope_kernel(unsigned short* __restrict__ qkv, const int* __restrict__ positions) {
  const int total = BATCH * S_LEN * (NH + NKV) * 64;
  int idx = blockIdx.x * blockDim.x + threadIdx.x;
  if (idx >= total) return;
  int i    = idx & 63;
  int t    = idx >> 6;
  int head = t % (NH + NKV);
  int bs   = t / (NH + NKV);
  int off  = (head < NH) ? head * HD : HIDDEN + (head - NH) * HD;
  unsigned short* p = qkv + (size_t)bs * QKV_LD + off;
  float pos = (float)positions[bs];
  float ang = pos * __expf(-(float)i * 0.14391156615f);   // 10000^(-i/64)
  float c = __cosf(ang), s = __sinf(ang);
  float x1 = bf2f(p[i]), x2 = bf2f(p[i + 64]);
  p[i]      = f2bf(x1 * c - x2 * s);
  p[i + 64] = f2bf(x2 * c + x1 * s);
}

// --------------------------- flash attention -------------------------------

__global__ __launch_bounds__(128)
void attn_kernel(const unsigned short* __restrict__ qkv, unsigned short* __restrict__ ctx) {
  __shared__ unsigned short Ktile[32 * HD];     // [key][dim]      8 KB
  __shared__ unsigned short VTtile[HD * 32];    // [dim][key]      8 KB
  __shared__ unsigned short Ptile[4][16 * 32];  // per-wave P      4 KB

  const int b    = blockIdx.z;
  const int h    = blockIdx.y;
  const int qblk = blockIdx.x;
  const int kvh  = h >> 2;

  const int tid  = threadIdx.x;
  const int lane = tid & 31;
  const int wave = tid >> 5;
  const int half = lane >> 4;
  const int l16  = lane & 15;

  const int qbase = qblk * 64 + wave * 16;
  const size_t rowbase = (size_t)b * S_LEN;
  const int koff = HIDDEN + kvh * HD;
  const int voff = HIDDEN + NKV * HD + kvh * HD;

  BF16Frag qf[4];
#pragma unroll
  for (int d = 0; d < 4; ++d) {
    const unsigned short* p = qkv + (rowbase + qbase + l16) * QKV_LD + h * HD + d * 32;
    qf[d].q[0] = *(const uint4*)(p + half * 8);
    qf[d].q[1] = *(const uint4*)(p + 16 + half * 8);
  }

  v8f acc[8];
  float m[8], l[8];
#pragma unroll
  for (int ni = 0; ni < 8; ++ni)
#pragma unroll
    for (int r = 0; r < 8; ++r) acc[ni][r] = 0.f;
#pragma unroll
  for (int r = 0; r < 8; ++r) { m[r] = -1e30f; l[r] = 0.f; }

  const int nkb = qblk * 2 + 2;
  const float scale = 0.08838834764831845f;

  for (int kb = 0; kb < nkb; ++kb) {
    __syncthreads();
    {   // stage K (async) + V (manual transposed store)
      int key = tid >> 2, ch = tid & 3;
      size_t srow = (rowbase + kb * 32 + key) * QKV_LD;
      const unsigned short* kp = qkv + srow + koff + ch * 32;
      unsigned short* kd = &Ktile[key * HD + ch * 32];
#pragma unroll
      for (int j = 0; j < 4; ++j) async_copy16(kp + j * 8, kd + j * 8);
      union { uint4 q[4]; unsigned short s[32]; } vv;
      const uint4* vp = (const uint4*)(qkv + srow + voff + ch * 32);
#pragma unroll
      for (int j = 0; j < 4; ++j) vv.q[j] = vp[j];
#pragma unroll
      for (int j = 0; j < 32; ++j) VTtile[(ch * 32 + j) * 32 + key] = vv.s[j];
      wait_async<0>();
    }
    __syncthreads();

    // ---- scores S[16q x 32k]: two 16x16 tiles, 4 WMMAs each over head dim
    v8f st[2];
#pragma unroll
    for (int kk = 0; kk < 2; ++kk)
#pragma unroll
      for (int r = 0; r < 8; ++r) st[kk][r] = 0.f;
#pragma unroll
    for (int kk = 0; kk < 2; ++kk)
#pragma unroll
      for (int d = 0; d < 4; ++d) {
        BF16Frag kf;
        const unsigned short* p = &Ktile[(kk * 16 + l16) * HD + d * 32];
        kf.q[0] = *(const uint4*)(p + half * 8);
        kf.q[1] = *(const uint4*)(p + 16 + half * 8);
        st[kk] = wmma_bf16(qf[d], kf, st[kk]);
      }

    // ---- scale + causal mask
#pragma unroll
    for (int kk = 0; kk < 2; ++kk)
#pragma unroll
      for (int r = 0; r < 8; ++r) {
        int key = kb * 32 + kk * 16 + l16;
        int qrow = qbase + r + 8 * half;
        float sc = st[kk][r] * scale;
        st[kk][r] = (key <= qrow) ? sc : -1e30f;
      }

    // ---- online softmax (row reductions across 16-lane halves)
    float bm[8];
#pragma unroll
    for (int r = 0; r < 8; ++r) bm[r] = fmaxf(st[0][r], st[1][r]);
#pragma unroll
    for (int off = 8; off > 0; off >>= 1)
#pragma unroll
      for (int r = 0; r < 8; ++r)
        bm[r] = fmaxf(bm[r], __shfl_xor(bm[r], off, 16));

    float alpha[8];
#pragma unroll
    for (int r = 0; r < 8; ++r) {
      float mn = fmaxf(m[r], bm[r]);
      alpha[r] = __expf(m[r] - mn);
      m[r] = mn;
    }

    float ps[8];
#pragma unroll
    for (int r = 0; r < 8; ++r) ps[r] = 0.f;
#pragma unroll
    for (int kk = 0; kk < 2; ++kk)
#pragma unroll
      for (int r = 0; r < 8; ++r) {
        float pv = __expf(st[kk][r] - m[r]);
        st[kk][r] = pv;
        ps[r] += pv;
      }
#pragma unroll
    for (int off = 8; off > 0; off >>= 1)
#pragma unroll
      for (int r = 0; r < 8; ++r)
        ps[r] += __shfl_xor(ps[r], off, 16);
#pragma unroll
    for (int r = 0; r < 8; ++r) l[r] = l[r] * alpha[r] + ps[r];

#pragma unroll
    for (int ni = 0; ni < 8; ++ni)
#pragma unroll
      for (int r = 0; r < 8; ++r) acc[ni][r] *= alpha[r];

    // ---- P (C-layout) -> bf16 A-fragment via per-wave LDS transpose
    unsigned short* pw = &Ptile[wave][0];
#pragma unroll
    for (int kk = 0; kk < 2; ++kk)
#pragma unroll
      for (int r = 0; r < 8; ++r)
        pw[(r + 8 * half) * 32 + kk * 16 + l16] = f2bf(st[kk][r]);
    asm volatile("s_wait_dscnt 0" ::: "memory");

    BF16Frag pf;
    {
      const unsigned short* p = pw + l16 * 32;
      pf.q[0] = *(const uint4*)(p + half * 8);
      pf.q[1] = *(const uint4*)(p + 16 + half * 8);
    }

    // ---- O += P . V
#pragma unroll
    for (int ni = 0; ni < 8; ++ni) {
      BF16Frag vf;
      const unsigned short* p = &VTtile[(ni * 16 + l16) * 32];
      vf.q[0] = *(const uint4*)(p + half * 8);
      vf.q[1] = *(const uint4*)(p + 16 + half * 8);
      acc[ni] = wmma_bf16(pf, vf, acc[ni]);
    }
  }

#pragma unroll
  for (int ni = 0; ni < 8; ++ni)
#pragma unroll
    for (int r = 0; r < 8; ++r) {
      int qrow = qbase + r + 8 * half;
      ctx[(rowbase + qrow) * HIDDEN + h * HD + ni * 16 + l16] = f2bf(acc[ni][r] / l[r]);
    }
}

// ------------------------------ launcher -----------------------------------

extern "C" void kernel_launch(void* const* d_in, const int* in_sizes, int n_in,
                              void* d_out, int out_size, void* d_ws, size_t ws_size,
                              hipStream_t stream) {
  const int*   positions = (const int*)d_in[0];
  const float* hidden    = (const float*)d_in[1];
  const float* w_qkv     = (const float*)d_in[2];
  const float* w_o       = (const float*)d_in[3];
  float* out = (float*)d_out;

  unsigned short* hbf   = (unsigned short*)d_ws;              // 16,777,216
  unsigned short* wqkvT = hbf   + (size_t)M_ROWS * HIDDEN;    // 25,165,824
  unsigned short* woT   = wqkvT + (size_t)HIDDEN * QKV_LD;    // 16,777,216
  unsigned short* qkvb  = woT   + (size_t)HIDDEN * HIDDEN;    // 25,165,824
  unsigned short* ctx   = hbf;                                // reuse

  cvt_bf16_kernel<<<(M_ROWS * HIDDEN / 4 + 255) / 256, 256, 0, stream>>>(
      hidden, hbf, M_ROWS * HIDDEN / 4);
  cvt_transpose_kernel<<<dim3(QKV_LD / 32, HIDDEN / 32), 256, 0, stream>>>(
      w_qkv, wqkvT, HIDDEN, QKV_LD);
  cvt_transpose_kernel<<<dim3(HIDDEN / 32, HIDDEN / 32), 256, 0, stream>>>(
      w_o, woT, HIDDEN, HIDDEN);

  gemm_bf16<false><<<dim3(QKV_LD / 128, M_ROWS / 256), 256, 0, stream>>>(
      hbf, wqkvT, qkvb, M_ROWS, QKV_LD, HIDDEN);

  {
    int total = BATCH * S_LEN * (NH + NKV) * 64;
    rope_kernel<<<(total + 255) / 256, 256, 0, stream>>>(qkvb, positions);
  }

  attn_kernel<<<dim3(S_LEN / 64, NH, BATCH), 128, 0, stream>>>(qkvb, ctx);

  gemm_bf16<true><<<dim3(HIDDEN / 128, M_ROWS / 256), 256, 0, stream>>>(
      ctx, woT, out, M_ROWS, HIDDEN, HIDDEN);
}
